// MultiHeadAttention_87849261073072
// MI455X (gfx1250) — matrix operand order repaired and checked by reading, not compiled, not run on hardware
//
#include <hip/hip_runtime.h>

typedef __bf16 bf16;
typedef __attribute__((ext_vector_type(16))) __bf16 v16bf;
typedef __attribute__((ext_vector_type(8)))  float  v8f;

#define NUM_HEADS 16
#define D_IN   1024
#define D_OUT  1024
#define D_HEAD 64
#define BATCH  4
#define SEQ    2048

// ---------------------------------------------------------------------------
// Async global->LDS copy path (CDNA5 GLOBAL_LOAD_ASYNC_TO_LDS_B128, ASYNCcnt).
// Prototype (from toolchain): (int4 as1*, int4 as3*, imm offset, imm cpol).
#if defined(__gfx1250__) &&                                                   \
    __has_builtin(__builtin_amdgcn_global_load_async_to_lds_b128) &&          \
    __has_builtin(__builtin_amdgcn_s_wait_asynccnt)
#define USE_ASYNC_LDS 1
typedef __attribute__((ext_vector_type(4))) int i32x4;
typedef __attribute__((address_space(1))) i32x4 i32x4_g;
typedef __attribute__((address_space(3))) i32x4 i32x4_l;
__device__ __forceinline__ void async_copy_b128(const void* g, const void* l) {
  // generic global pointer == as(1) address; generic LDS pointer low 32 bits
  // == LDS byte offset (ISA 10.2 aperture rules).
  __builtin_amdgcn_global_load_async_to_lds_b128(
      (i32x4_g*)(unsigned long long)g,
      (i32x4_l*)(unsigned int)(unsigned long long)l, 0, 0);
}
__device__ __forceinline__ void async_wait0() {
  __builtin_amdgcn_s_wait_asynccnt(0);
}
#else
#define USE_ASYNC_LDS 0
#endif

// ---------------------------------------------------------------------------
// WMMA fragment loaders (wave32 layouts per CDNA5 ISA 7.12.2).
// A (16x32 bf16): lane r=lane&15 holds row r; lanes 0-15 carry K=[0..7]+[16..23],
// lanes 16-31 carry K=[8..15]+[24..31]; each half is one contiguous 16B run.
__device__ __forceinline__ v16bf load_frag_a(const bf16* base, int stride, int kOff) {
  const int lane = threadIdx.x & 31;
  const int row  = lane & 15;
  const int h    = (lane >> 4) * 8;
  const bf16* p  = base + row * stride + kOff + h;
  union { v16bf v; uint4 u[2]; } r;
  r.u[0] = *(const uint4*)(p);
  r.u[1] = *(const uint4*)(p + 16);
  return r.v;
}
// B (32x16 bf16) loaded from a K-contiguous transposed tile BT[n][k]:
// lane n=lane&15 holds column n; lanes 0-15 carry K=[0..7]+[16..23], 16-31 K=[8..15]+[24..31].
__device__ __forceinline__ v16bf load_frag_b(const bf16* baseT, int stride, int kOff) {
  const int lane = threadIdx.x & 31;
  const int col  = lane & 15;
  const int h    = (lane >> 4) * 8;
  const bf16* p  = baseT + col * stride + kOff + h;
  union { v16bf v; uint4 u[2]; } r;
  r.u[0] = *(const uint4*)(p);
  r.u[1] = *(const uint4*)(p + 16);
  return r.v;
}

__device__ __forceinline__ v8f wmma_bf16(v16bf a, v16bf b, v8f c) {
  // emits v_wmma_f32_16x16x32_bf16
  return __builtin_amdgcn_wmma_f32_16x16x32_bf16(false, a, false, b, (short)0, c,
                                                 false, false);
}

// ---------------------------------------------------------------------------
// Kernel 1: Y = x @ W for W in {Wq,Wk,Wv} (blockIdx.z selects), f32 in,
// bf16 out in (B,H,S,Dh) layout. 128x128 tile, 8 waves (4x2), 8 accum tiles/wave.
// Both tiles need f32->bf16 conversion, so staging stays on the VALU path.
__global__ __launch_bounds__(256) void qkv_gemm(
    const float* __restrict__ x, const float* __restrict__ Wq,
    const float* __restrict__ Wk, const float* __restrict__ Wv,
    bf16* __restrict__ Qp, bf16* __restrict__ Kp, bf16* __restrict__ Vp)
{
  const float* W = (blockIdx.z == 0) ? Wq : (blockIdx.z == 1) ? Wk : Wv;
  bf16* Y        = (blockIdx.z == 0) ? Qp : (blockIdx.z == 1) ? Kp : Vp;

  __shared__ __align__(16) bf16 lA[128 * 40];   // x tile  [m][k], stride 40 (pad)
  __shared__ __align__(16) bf16 lB[128 * 40];   // W tileT [n][k], stride 40

  const int m0 = blockIdx.y * 128;
  const int n0 = blockIdx.x * 128;
  const int tid = threadIdx.x;
  const int wave = tid >> 5;
  const int wm = (wave & 3) * 32;
  const int wn = (wave >> 2) * 64;

  v8f acc[2][4] = {};

  for (int k0 = 0; k0 < D_IN; k0 += 32) {
    __syncthreads();
    // stage A: 128x32 f32 -> bf16, coalesced float4 reads
    #pragma unroll
    for (int p = 0; p < 4; ++p) {
      int idx = p * 256 + tid;
      int r = idx >> 3, c = (idx & 7) * 4;
      float4 xv = *(const float4*)(x + (long)(m0 + r) * D_IN + k0 + c);
      bf16* d = lA + r * 40 + c;
      d[0] = (bf16)xv.x; d[1] = (bf16)xv.y; d[2] = (bf16)xv.z; d[3] = (bf16)xv.w;
    }
    // stage B transposed: W rows k0..k0+31, cols n0..n0+127 -> lB[n][k]
    #pragma unroll
    for (int p = 0; p < 4; ++p) {
      int idx = p * 256 + tid;
      int kk = idx >> 5, c = (idx & 31) * 4;
      float4 wv = *(const float4*)(W + (long)(k0 + kk) * D_OUT + n0 + c);
      lB[(c + 0) * 40 + kk] = (bf16)wv.x;
      lB[(c + 1) * 40 + kk] = (bf16)wv.y;
      lB[(c + 2) * 40 + kk] = (bf16)wv.z;
      lB[(c + 3) * 40 + kk] = (bf16)wv.w;
    }
    if (k0 + 32 < D_IN) {  // pull next k-tile toward L2/L1 (global_prefetch_b8)
      __builtin_prefetch(x + (long)(m0 + (tid >> 3)) * D_IN + k0 + 32, 0, 1);
      __builtin_prefetch(W + (long)(k0 + 32 + (tid >> 5)) * D_OUT + n0, 0, 1);
    }
    __syncthreads();
    v16bf a0 = load_frag_a(lA + (wm + 0) * 40, 40, 0);
    v16bf a1 = load_frag_a(lA + (wm + 16) * 40, 40, 0);
    #pragma unroll
    for (int nt = 0; nt < 4; ++nt) {
      v16bf bb = load_frag_b(lB + (wn + nt * 16) * 40, 40, 0);
      acc[0][nt] = wmma_bf16(a0, bb, acc[0][nt]);
      acc[1][nt] = wmma_bf16(a1, bb, acc[1][nt]);
    }
  }

  // epilogue: C layout (VGPR i -> row i / 8+i, lanes = cols) -> (B,H,S,Dh) bf16
  const int lane = tid & 31;
  const int lr = (lane >> 4) * 8;
  const int lc = lane & 15;
  #pragma unroll
  for (int mt = 0; mt < 2; ++mt)
    #pragma unroll
    for (int nt = 0; nt < 4; ++nt) {
      int gm = m0 + wm + mt * 16 + lr;
      int gn = n0 + wn + nt * 16 + lc;
      int b = gm >> 11, s = gm & 2047;
      int h = gn >> 6, dd = gn & 63;
      bf16* o = Y + (((long)(b * NUM_HEADS + h) * SEQ + s) * D_HEAD + dd);
      #pragma unroll
      for (int i = 0; i < 8; ++i) o[(long)i * D_HEAD] = (bf16)acc[mt][nt][i];
    }
}

// ---------------------------------------------------------------------------
// Kernel 2: causal flash attention. Block = (b, h, 128-query tile), 8 waves,
// each wave owns 16 q rows and the full Dh=64 output (4 f32 accum tiles).
// K tile is a pure bf16 copy -> async DMA to LDS; V tile needs a transpose,
// which overlaps with the in-flight async copies.
__global__ __launch_bounds__(256) void attn_kernel(
    const bf16* __restrict__ Qp, const bf16* __restrict__ Kp,
    const bf16* __restrict__ Vp, bf16* __restrict__ Ctx)
{
  __shared__ __align__(16) bf16 lK[32 * 72];      // K tile [key][d], stride 72
  __shared__ __align__(16) bf16 lV[64 * 40];      // V tileT [d][key], stride 40
  __shared__ __align__(16) bf16 lP[8 * 16 * 40];  // per-wave P tile [16][32]

  const int tid = threadIdx.x;
  const int wave = tid >> 5;
  const int lane = tid & 31;
  const int b = blockIdx.z, h = blockIdx.y;
  const int q0 = blockIdx.x * 128;
  const long bhS = (long)(b * NUM_HEADS + h) * SEQ;
  const int qw = q0 + wave * 16;

  // Q fragments stay in registers for the whole block
  v16bf qf0 = load_frag_a(Qp + (bhS + qw) * D_HEAD, D_HEAD, 0);
  v16bf qf1 = load_frag_a(Qp + (bhS + qw) * D_HEAD, D_HEAD, 32);

  v8f o[4] = {};
  float mrun[8], lrun[8];
  #pragma unroll
  for (int i = 0; i < 8; ++i) { mrun[i] = -1e30f; lrun[i] = 0.0f; }

  const int cgrp = lane >> 4;    // C-layout row group (0: rows 0-7, 1: rows 8-15)
  const int ccol = lane & 15;    // C-layout column
  bf16* myP = lP + wave * (16 * 40);

  // block-uniform causal bound -> __syncthreads stays legal
  for (int kt = 0; kt < q0 + 128; kt += 32) {
    __syncthreads();
#if USE_ASYNC_LDS
    // K tile: 32x64 bf16 = 256 x 16B chunks; one async b128 per thread,
    // per-lane LDS address keeps the padded stride-72 layout.
    {
      int key = tid >> 3, c = (tid & 7) * 8;
      async_copy_b128(Kp + (bhS + kt + key) * D_HEAD + c, lK + key * 72 + c);
    }
    // V tile transposed (needs data movement through VGPRs)
    #pragma unroll
    for (int p = 0; p < 2; ++p) {
      int idx = p * 256 + tid;
      int key = idx >> 4, c = (idx & 15) * 4;
      union { uint2 u; bf16 e[4]; } t;
      t.u = *(const uint2*)(Vp + (bhS + kt + key) * D_HEAD + c);
      lV[(c + 0) * 40 + key] = t.e[0];
      lV[(c + 1) * 40 + key] = t.e[1];
      lV[(c + 2) * 40 + key] = t.e[2];
      lV[(c + 3) * 40 + key] = t.e[3];
    }
    async_wait0();
#else
    #pragma unroll
    for (int p = 0; p < 2; ++p) {
      int idx = p * 256 + tid;
      int key = idx >> 4, c = (idx & 15) * 4;
      *(uint2*)(lK + key * 72 + c) =
          *(const uint2*)(Kp + (bhS + kt + key) * D_HEAD + c);
      union { uint2 u; bf16 e[4]; } t;
      t.u = *(const uint2*)(Vp + (bhS + kt + key) * D_HEAD + c);
      lV[(c + 0) * 40 + key] = t.e[0];
      lV[(c + 1) * 40 + key] = t.e[1];
      lV[(c + 2) * 40 + key] = t.e[2];
      lV[(c + 3) * 40 + key] = t.e[3];
    }
#endif
    __syncthreads();

    // scores: 16q x 32k = two 16x16 tiles, K-dim = Dh = 64 (2 WMMAs each)
    v8f sc0 = {}, sc1 = {};
    {
      v16bf b00 = load_frag_b(lK, 72, 0);
      v16bf b01 = load_frag_b(lK, 72, 32);
      sc0 = wmma_bf16(qf0, b00, sc0);
      sc0 = wmma_bf16(qf1, b01, sc0);
      v16bf b10 = load_frag_b(lK + 16 * 72, 72, 0);
      v16bf b11 = load_frag_b(lK + 16 * 72, 72, 32);
      sc1 = wmma_bf16(qf0, b10, sc1);
      sc1 = wmma_bf16(qf1, b11, sc1);
    }

    // scale (1/sqrt(64)), causal mask, online softmax (16-lane xor reductions)
    #pragma unroll
    for (int i = 0; i < 8; ++i) {
      int qg = qw + cgrp * 8 + i;
      float v0 = sc0[i] * 0.125f;
      float v1 = sc1[i] * 0.125f;
      v0 = ((kt + ccol) > qg) ? -1e30f : v0;
      v1 = ((kt + 16 + ccol) > qg) ? -1e30f : v1;
      float rm = fmaxf(v0, v1);
      #pragma unroll
      for (int off = 1; off < 16; off <<= 1) rm = fmaxf(rm, __shfl_xor(rm, off, 32));
      float mnew = fmaxf(mrun[i], rm);
      float alpha = __expf(mrun[i] - mnew);
      mrun[i] = mnew;
      v0 = __expf(v0 - mnew);
      v1 = __expf(v1 - mnew);
      float rs = v0 + v1;
      #pragma unroll
      for (int off = 1; off < 16; off <<= 1) rs += __shfl_xor(rs, off, 32);
      lrun[i] = lrun[i] * alpha + rs;
      #pragma unroll
      for (int nt = 0; nt < 4; ++nt) o[nt][i] *= alpha;
      sc0[i] = v0;
      sc1[i] = v1;
    }

    __syncthreads();
    // re-materialize P in A-fragment layout via per-wave LDS tile
    #pragma unroll
    for (int i = 0; i < 8; ++i) {
      myP[(cgrp * 8 + i) * 40 + ccol]      = (bf16)sc0[i];
      myP[(cgrp * 8 + i) * 40 + 16 + ccol] = (bf16)sc1[i];
    }
    __syncthreads();

    v16bf pf = load_frag_a(myP, 40, 0);
    #pragma unroll
    for (int nt = 0; nt < 4; ++nt) {
      v16bf vf = load_frag_b(lV + (nt * 16) * 40, 40, 0);
      o[nt] = wmma_bf16(pf, vf, o[nt]);
    }
  }

  // normalize and store ctx in (B, S, H*Dh) bf16 layout for the output GEMM
  #pragma unroll
  for (int nt = 0; nt < 4; ++nt) {
    int cc = h * D_HEAD + nt * 16 + ccol;
    bf16* out = Ctx + ((long)b * SEQ + qw + cgrp * 8) * D_OUT + cc;
    #pragma unroll
    for (int i = 0; i < 8; ++i)
      out[(long)i * D_OUT] = (bf16)(o[nt][i] / lrun[i]);
  }
}

// ---------------------------------------------------------------------------
// Kernel 3: out = ctx @ Wo + bo. A tile is already bf16 -> async DMA to LDS;
// Wo tile needs f32->bf16 transpose-convert which overlaps with the DMA.
__global__ __launch_bounds__(256) void out_gemm(
    const bf16* __restrict__ Ctx, const float* __restrict__ Wo,
    const float* __restrict__ bo, float* __restrict__ Y)
{
  __shared__ __align__(16) bf16 lA[128 * 40];
  __shared__ __align__(16) bf16 lB[128 * 40];
  const int m0 = blockIdx.y * 128, n0 = blockIdx.x * 128;
  const int tid = threadIdx.x, wave = tid >> 5;
  const int wm = (wave & 3) * 32, wn = (wave >> 2) * 64;
  v8f acc[2][4] = {};

  for (int k0 = 0; k0 < D_OUT; k0 += 32) {
    __syncthreads();
#if USE_ASYNC_LDS
    // A tile: 128x32 bf16 = 512 x 16B chunks, 2 async b128 per thread
    #pragma unroll
    for (int p = 0; p < 2; ++p) {
      int idx = p * 256 + tid;
      int r = idx >> 2, c = (idx & 3) * 8;
      async_copy_b128(Ctx + (long)(m0 + r) * D_OUT + k0 + c, lA + r * 40 + c);
    }
#else
    #pragma unroll
    for (int p = 0; p < 4; ++p) {
      int idx = p * 256 + tid;
      int r = idx >> 3, c = (idx & 7) * 4;
      *(uint2*)(lA + r * 40 + c) =
          *(const uint2*)(Ctx + (long)(m0 + r) * D_OUT + k0 + c);
    }
#endif
    #pragma unroll
    for (int p = 0; p < 4; ++p) {
      int idx = p * 256 + tid;
      int kk = idx >> 5, c = (idx & 31) * 4;
      float4 wv = *(const float4*)(Wo + (long)(k0 + kk) * D_OUT + n0 + c);
      lB[(c + 0) * 40 + kk] = (bf16)wv.x;
      lB[(c + 1) * 40 + kk] = (bf16)wv.y;
      lB[(c + 2) * 40 + kk] = (bf16)wv.z;
      lB[(c + 3) * 40 + kk] = (bf16)wv.w;
    }
    if (k0 + 32 < D_OUT)
      __builtin_prefetch(Wo + (long)(k0 + 32 + (tid >> 5)) * D_OUT + n0, 0, 1);
#if USE_ASYNC_LDS
    async_wait0();
#endif
    __syncthreads();
    v16bf a0 = load_frag_a(lA + (wm + 0) * 40, 40, 0);
    v16bf a1 = load_frag_a(lA + (wm + 16) * 40, 40, 0);
    #pragma unroll
    for (int nt = 0; nt < 4; ++nt) {
      v16bf bb = load_frag_b(lB + (wn + nt * 16) * 40, 40, 0);
      acc[0][nt] = wmma_bf16(a0, bb, acc[0][nt]);
      acc[1][nt] = wmma_bf16(a1, bb, acc[1][nt]);
    }
  }

  const int lane = tid & 31;
  const int lr = (lane >> 4) * 8, lc = lane & 15;
  #pragma unroll
  for (int mt = 0; mt < 2; ++mt)
    #pragma unroll
    for (int nt = 0; nt < 4; ++nt) {
      int gm = m0 + wm + mt * 16 + lr;
      int gn = n0 + wn + nt * 16 + lc;
      float bias = bo[gn];
      float* out = Y + (long)gm * D_OUT + gn;
      #pragma unroll
      for (int i = 0; i < 8; ++i) out[(long)i * D_OUT] = acc[mt][nt][i] + bias;
    }
}

// ---------------------------------------------------------------------------
extern "C" void kernel_launch(void* const* d_in, const int* in_sizes, int n_in,
                              void* d_out, int out_size, void* d_ws, size_t ws_size,
                              hipStream_t stream) {
  (void)in_sizes; (void)n_in; (void)out_size; (void)ws_size;
  const float* x  = (const float*)d_in[0];
  const float* Wq = (const float*)d_in[1];
  const float* Wk = (const float*)d_in[2];
  const float* Wv = (const float*)d_in[3];
  const float* Wo = (const float*)d_in[4];
  const float* bo = (const float*)d_in[5];
  float* out = (float*)d_out;

  // workspace: Q,K,V in (B,H,S,Dh) bf16 + ctx in (B,S,H*Dh) bf16  (64 MB total)
  const size_t tok = (size_t)BATCH * SEQ * D_OUT;  // 8Mi elements
  bf16* Qp  = (bf16*)d_ws;
  bf16* Kp  = Qp + tok;
  bf16* Vp  = Kp + tok;
  bf16* Cp  = Vp + tok;

  dim3 blk(256);
  qkv_gemm<<<dim3(D_OUT / 128, (BATCH * SEQ) / 128, 3), blk, 0, stream>>>(
      x, Wq, Wk, Wv, Qp, Kp, Vp);
  attn_kernel<<<dim3(SEQ / 128, NUM_HEADS, BATCH), blk, 0, stream>>>(
      Qp, Kp, Vp, Cp);
  out_gemm<<<dim3(D_OUT / 128, (BATCH * SEQ) / 128), blk, 0, stream>>>(
      Cp, Wo, bo, out);
}
